// newmodel_12154757448430
// MI455X (gfx1250) — matrix operand order repaired and checked by reading, not compiled, not run on hardware
//
#include <hip/hip_runtime.h>
#include <hip/hip_bf16.h>
#include <math.h>

typedef __attribute__((ext_vector_type(16))) _Float16 v16h;
typedef __attribute__((ext_vector_type(8)))  float    v8f;

#define TLD 36   /* 32 + 4 floats pad: rows 144B apart -> distinct banks */

// ---------------------------------------------------------------------------
// WMMA fragment helpers (wave32, 16x16x32 f16 -> f32). LDS tiles hold f32,
// conversion to f16 happens at fragment-load time.
// ---------------------------------------------------------------------------
__device__ __forceinline__ v16h frag_a_f32(const float* As, int mrow_base, int lane) {
  // A 16x32 layout: lanes 0-15 row M=lane, K={0..7,16..23};
  // lanes 16-31 row M=lane-16, K={8..15,24..31}
  const float* p = As + (mrow_base + (lane & 15)) * TLD;
  int kb = (lane & 16) ? 8 : 0;
  v16h a;
#pragma unroll
  for (int i = 0; i < 8; ++i) {
    a[i] = (_Float16)p[kb + i];
    a[8 + i] = (_Float16)p[16 + kb + i];
  }
  return a;
}

__device__ __forceinline__ v16h frag_b_f32(const float* Bs, int ncol_base, int lane) {
  // B 32x16 layout: lanes 0-15 col N=lane K={0..7,16..23}; lanes 16-31 col N=lane-16 K={8..15,24..31}
  const float* p = Bs + (ncol_base + (lane & 15)) * TLD;
  int k0 = (lane & 16) ? 8 : 0;
  v16h b;
#pragma unroll
  for (int i = 0; i < 8; ++i) {
    b[i] = (_Float16)p[k0 + i];
    b[8 + i] = (_Float16)p[16 + k0 + i];
  }
  return b;
}

__device__ __forceinline__ v8f wmma_step(v16h a, v16h b, v8f c) {
  return __builtin_amdgcn_wmma_f32_16x16x32_f16(false, a, false, b, (short)0, c,
                                                false, false);
}

__device__ __forceinline__ float activate(float v, int act) {
  if (act == 1) return v > 0.f ? v : 0.f;                         // relu
  if (act == 2) return (v > 20.f) ? v : log1pf(__expf(v));        // softplus
  if (act == 3) return v / (1.f + __expf(-v));                    // silu
  return v;
}

// Async copy: one 64x32-f32 tile (row-major, TLD-padded) from a dense
// [rows x ldg] f32 matrix. 128 threads, 4 x b128 per lane, all in-bounds.
__device__ __forceinline__ void tile_async_fill(float* dst_lds, const float* src,
                                                long row0, long ldg, long col0,
                                                int tid) {
#pragma unroll
  for (int i = 0; i < 4; ++i) {
    int g = i * 128 + tid;                 // 512 groups of 4 floats
    int row = g >> 3, c4 = (g & 7) * 4;
    const float* gp = src + (row0 + row) * ldg + col0 + c4;
    unsigned lds = (unsigned)(uintptr_t)(&dst_lds[row * TLD + c4]);
    asm volatile("global_load_async_to_lds_b128 %0, %1, off"
                 :: "v"(lds), "v"(gp) : "memory");
  }
}

__device__ __forceinline__ void wait_async_lds() {
  asm volatile("s_wait_asynccnt 0x0" ::: "memory");
}

// ---------------------------------------------------------------------------
// GEMM: out[N][M] = X[N][K(ldx)] * W[M][K]^T  (+bias[M], activation)
// ---------------------------------------------------------------------------
__global__ __launch_bounds__(128)
void gemm_xwT_wmma(const float* __restrict__ W, const float* __restrict__ X,
                   const float* __restrict__ bias, float* __restrict__ out,
                   int M, int N, int K, int ldx, int act) {
  __shared__ float Af[64 * TLD];
  __shared__ float Bf[64 * TLD];
  int tid = threadIdx.x;
  int lane = tid & 31, wave = tid >> 5;
  int wm = (wave >> 1) * 32, wn = (wave & 1) * 32;
  int m0 = blockIdx.y * 64, n0 = blockIdx.x * 64;

  v8f acc[2][2];
#pragma unroll
  for (int i = 0; i < 2; ++i)
#pragma unroll
    for (int j = 0; j < 2; ++j)
#pragma unroll
      for (int r = 0; r < 8; ++r) acc[i][j][r] = 0.f;

  for (int k0 = 0; k0 < K; k0 += 32) {
    bool fullK = (k0 + 32 <= K);
    if (fullK && (m0 + 64 <= M)) {
      tile_async_fill(Af, W, m0, K, k0, tid);       // weights, dense rows
    } else {
#pragma unroll
      for (int i = 0; i < 16; ++i) {                // branchless clamped fill
        int e = i * 128 + tid;
        int mm = e >> 5, kk = e & 31;
        int m = m0 + mm, k = k0 + kk;
        int mc = m < M ? m : M - 1;
        int kc = k < K ? k : K - 1;
        float v = W[(long)mc * K + kc];
        Af[mm * TLD + kk] = (m < M && k < K) ? v : 0.f;
      }
    }
    if (fullK && (n0 + 64 <= N)) {
      tile_async_fill(Bf, X, n0, ldx, k0, tid);     // activations, token rows
    } else {
#pragma unroll
      for (int i = 0; i < 16; ++i) {
        int e = i * 128 + tid;
        int nn = e >> 5, kk = e & 31;
        int n = n0 + nn, k = k0 + kk;
        int nc = n < N ? n : N - 1;
        int kc = k < K ? k : K - 1;
        float v = X[(long)nc * ldx + kc];
        Bf[nn * TLD + kk] = (n < N && k < K) ? v : 0.f;
      }
    }
    wait_async_lds();
    __syncthreads();
    v16h a0 = frag_a_f32(Af, wm, lane);
    v16h a1 = frag_a_f32(Af, wm + 16, lane);
    v16h b0 = frag_b_f32(Bf, wn, lane);
    v16h b1 = frag_b_f32(Bf, wn + 16, lane);
    acc[0][0] = wmma_step(a0, b0, acc[0][0]);
    acc[0][1] = wmma_step(a0, b1, acc[0][1]);
    acc[1][0] = wmma_step(a1, b0, acc[1][0]);
    acc[1][1] = wmma_step(a1, b1, acc[1][1]);
    __syncthreads();
  }

  int nlo = lane & 15, mb = (lane & 16) ? 8 : 0;
#pragma unroll
  for (int im = 0; im < 2; ++im)
#pragma unroll
    for (int jn = 0; jn < 2; ++jn)
#pragma unroll
      for (int r = 0; r < 8; ++r) {
        int m = m0 + wm + im * 16 + mb + r;
        int n = n0 + wn + jn * 16 + nlo;
        if (m < M && n < N) {
          float v = acc[im][jn][r];
          if (bias) v += bias[m];
          out[(long)n * M + m] = activate(v, act);
        }
      }
}

// ---------------------------------------------------------------------------
// Implicit-GEMM conv (NCHW, B=1). ks=1/3, stride 1/2 SAME, or transpose
// (lhs_dilation=2, pad=2). out[Cout][Hout*Wout], fused bias+act.
// ---------------------------------------------------------------------------
__global__ __launch_bounds__(128)
void conv_wmma(const float* __restrict__ X, const float* __restrict__ W,
               const float* __restrict__ bias, float* __restrict__ out,
               int Cin, int Cout, int Hin, int Win, int Hout, int Wout,
               int ks, int stride, int pad, int trans, int act) {
  const int K = Cin * ks * ks;
  const int N = Hout * Wout;
  const int kss = ks * ks;
  __shared__ float Af[64 * TLD];
  __shared__ float Bf[64 * TLD];
  int tid = threadIdx.x;
  int lane = tid & 31, wave = tid >> 5;
  int wm = (wave >> 1) * 32, wn = (wave & 1) * 32;
  int m0 = blockIdx.y * 64, n0 = blockIdx.x * 64;

  v8f acc[2][2];
#pragma unroll
  for (int i = 0; i < 2; ++i)
#pragma unroll
    for (int j = 0; j < 2; ++j)
#pragma unroll
      for (int r = 0; r < 8; ++r) acc[i][j][r] = 0.f;

  for (int k0 = 0; k0 < K; k0 += 32) {
    if ((k0 + 32 <= K) && (m0 + 64 <= Cout)) {
      tile_async_fill(Af, W, m0, K, k0, tid);       // weights [Cout][K]
    } else {
#pragma unroll
      for (int i = 0; i < 16; ++i) {
        int e = i * 128 + tid;
        int mm = e >> 5, kk = e & 31;
        int m = m0 + mm, k = k0 + kk;
        int mc = m < Cout ? m : Cout - 1;
        int kc = k < K ? k : K - 1;
        float v = W[(long)mc * K + kc];
        Af[mm * TLD + kk] = (m < Cout && k < K) ? v : 0.f;
      }
    }
    // B: gathered input patches, rows = output positions (branchless)
#pragma unroll
    for (int i = 0; i < 16; ++i) {
      int e = i * 128 + tid;
      int nn = e >> 5, kk = e & 31;
      int n = n0 + nn, k = k0 + kk;
      int nc = n < N ? n : N - 1;
      int kc = k < K ? k : K - 1;
      int cin = kc / kss, rem = kc % kss;
      int ky = rem / ks, kx = rem % ks;
      int oy = nc / Wout, ox = nc % Wout;
      bool ok = (n < N) && (k < K);
      int iy, ix;
      if (!trans) {
        iy = oy * stride + ky - pad;
        ix = ox * stride + kx - pad;
      } else {
        int dy = oy + ky - pad, dx = ox + kx - pad;
        ok = ok && ((dy & 1) == 0) && ((dx & 1) == 0) && (dy >= 0) && (dx >= 0);
        iy = dy >> 1;
        ix = dx >> 1;
      }
      ok = ok && (iy >= 0) && (iy < Hin) && (ix >= 0) && (ix < Win);
      int iyc = iy < 0 ? 0 : (iy >= Hin ? Hin - 1 : iy);
      int ixc = ix < 0 ? 0 : (ix >= Win ? Win - 1 : ix);
      float v = X[((long)cin * Hin + iyc) * Win + ixc];
      Bf[nn * TLD + kk] = ok ? v : 0.f;
    }
    wait_async_lds();
    __syncthreads();
    v16h a0 = frag_a_f32(Af, wm, lane);
    v16h a1 = frag_a_f32(Af, wm + 16, lane);
    v16h b0 = frag_b_f32(Bf, wn, lane);
    v16h b1 = frag_b_f32(Bf, wn + 16, lane);
    acc[0][0] = wmma_step(a0, b0, acc[0][0]);
    acc[0][1] = wmma_step(a0, b1, acc[0][1]);
    acc[1][0] = wmma_step(a1, b0, acc[1][0]);
    acc[1][1] = wmma_step(a1, b1, acc[1][1]);
    __syncthreads();
  }

  int nlo = lane & 15, mb = (lane & 16) ? 8 : 0;
#pragma unroll
  for (int im = 0; im < 2; ++im)
#pragma unroll
    for (int jn = 0; jn < 2; ++jn)
#pragma unroll
      for (int r = 0; r < 8; ++r) {
        int m = m0 + wm + im * 16 + mb + r;
        int n = n0 + wn + jn * 16 + nlo;
        if (m < Cout && n < N) {
          float v = acc[im][jn][r] + bias[m];
          out[(long)m * N + n] = activate(v, act);
        }
      }
}

// ---------------------------------------------------------------------------
// Elementwise / data-movement kernels
// ---------------------------------------------------------------------------
__global__ void nhwc_to_nchw_k(const float* __restrict__ x, float* __restrict__ o,
                               int H, int W, int C) {
  long idx = (long)blockIdx.x * blockDim.x + threadIdx.x;
  long n = (long)H * W * C;
  if (idx >= n) return;
  int c = idx % C; long t = idx / C;
  int xw = t % W, y = t / W;
  o[((long)c * H + y) * W + xw] = x[idx];
}

__global__ void layernorm_c_k(const float* __restrict__ x, const float* __restrict__ g,
                              const float* __restrict__ b, float* __restrict__ y,
                              int C, int HW) {
  int t = blockIdx.x * blockDim.x + threadIdx.x;
  if (t >= HW) return;
  float m = 0.f;
  for (int c = 0; c < C; ++c) m += x[(long)c * HW + t];
  m /= C;
  float v = 0.f;
  for (int c = 0; c < C; ++c) { float d = x[(long)c * HW + t] - m; v += d * d; }
  v /= C;
  float r = rsqrtf(v + 1e-5f);
  for (int c = 0; c < C; ++c)
    y[(long)c * HW + t] = (x[(long)c * HW + t] - m) * r * g[c] + b[c];
}

// mode 0: out[i1][i0][i2]; mode 1: out[i2][i0][i1]; mode 2: out[i1][i2][i0]
__global__ void permute3_k(const float* __restrict__ in, float* __restrict__ out,
                           int d0, int d1, int d2, int mode) {
  long idx = (long)blockIdx.x * blockDim.x + threadIdx.x;
  long n = (long)d0 * d1 * d2;
  if (idx >= n) return;
  int i2 = idx % d2; long t = idx / d2;
  int i1 = t % d1, i0 = t / d1;
  long o;
  if (mode == 0)      o = ((long)i1 * d0 + i0) * d2 + i2;
  else if (mode == 1) o = ((long)i2 * d0 + i0) * d1 + i1;
  else                o = ((long)i1 * d2 + i2) * d0 + i0;
  out[o] = in[idx];
}

__global__ void pixshuf_k(const float* __restrict__ in, float* __restrict__ out,
                          int Cin, int H, int W) {  // out: (Cin/4, 2H, 2W)
  int Co = Cin >> 2, Ho = H * 2, Wo = W * 2;
  long idx = (long)blockIdx.x * blockDim.x + threadIdx.x;
  long n = (long)Co * Ho * Wo;
  if (idx >= n) return;
  int ox = idx % Wo; long t = idx / Wo;
  int oy = t % Ho, c = t / Ho;
  int ci = c * 4 + (oy & 1) * 2 + (ox & 1);
  out[idx] = in[((long)ci * H + (oy >> 1)) * W + (ox >> 1)];
}

__global__ void pixunshuf_k(const float* __restrict__ in, float* __restrict__ out,
                            int Cin, int H, int W) {  // out: (4Cin, H/2, W/2)
  int Co = Cin * 4, Ho = H >> 1, Wo = W >> 1;
  long idx = (long)blockIdx.x * blockDim.x + threadIdx.x;
  long n = (long)Co * Ho * Wo;
  if (idx >= n) return;
  int x = idx % Wo; long t = idx / Wo;
  int y = t % Ho, c = t / Ho;
  out[idx] = in[((long)(c >> 2) * H + (y * 2 + ((c >> 1) & 1))) * W + (x * 2 + (c & 1))];
}

__global__ void add2_k(const float* a, const float* b, float* o, long n) {
  long i = (long)blockIdx.x * blockDim.x + threadIdx.x;
  if (i < n) o[i] = a[i] + b[i];
}
__global__ void add3_k(const float* a, const float* b, const float* c, float* o, long n) {
  long i = (long)blockIdx.x * blockDim.x + threadIdx.x;
  if (i < n) o[i] = a[i] + b[i] + c[i];
}

// depthwise causal conv1d (width 4) + bias + silu.  XZ token-major [B*L][1024]
__global__ void mamba_conv1d_k(const float* __restrict__ XZ, const float* __restrict__ cw,
                               const float* __restrict__ cb, float* __restrict__ U,
                               int Bm, int L, int D, int DXZ) {
  long idx = (long)blockIdx.x * blockDim.x + threadIdx.x;
  long n = (long)Bm * L * D;
  if (idx >= n) return;
  int d = idx % D; long t = idx / D;
  int l = t % L, b = t / L;
  float acc = cb[d];
#pragma unroll
  for (int k = 0; k < 4; ++k) {
    int ls = l - 3 + k;
    if (ls >= 0) acc += cw[d * 4 + k] * XZ[((long)b * L + ls) * DXZ + d];
  }
  U[idx] = acc / (1.f + __expf(-acc));
}

// selective scan: thread per (b,d). DT holds dt on entry, y_mod on exit.
__global__ void mamba_scan_k(const float* __restrict__ DBL, const float* __restrict__ U,
                             float* __restrict__ DT, const float* __restrict__ XZ,
                             const float* __restrict__ A_log, const float* __restrict__ Dp,
                             int Bm, int L) {
  int id = blockIdx.x * blockDim.x + threadIdx.x;
  if (id >= Bm * 512) return;
  int d = id % 512, b = id / 512;
  float A[16], h[16];
#pragma unroll
  for (int nn = 0; nn < 16; ++nn) { A[nn] = -__expf(A_log[d * 16 + nn]); h[nn] = 0.f; }
  float Dd = Dp[d];
  for (int l = 0; l < L; ++l) {
    long tok = (long)b * L + l;
    float dtv = DT[tok * 512 + d];
    float ut = U[tok * 512 + d];
    float du = dtv * ut, y = 0.f;
    const float* row = DBL + tok * 48;
#pragma unroll
    for (int nn = 0; nn < 16; ++nn) {
      h[nn] = __expf(dtv * A[nn]) * h[nn] + du * row[16 + nn];
      y += h[nn] * row[32 + nn];
    }
    float z = XZ[tok * 1024 + 512 + d];
    DT[tok * 512 + d] = (y + ut * Dd) * (z / (1.f + __expf(-z)));
  }
}

__global__ void chw_to_tc_k(const float* __restrict__ in, float* __restrict__ out,
                            int C, int HW) {
  long idx = (long)blockIdx.x * blockDim.x + threadIdx.x;
  if (idx >= (long)C * HW) return;
  int t = idx % HW, c = idx / HW;
  out[(long)t * C + c] = in[idx];
}

__global__ void l2norm_rows_k(const float* __restrict__ in, float* __restrict__ out,
                              int R, int C) {
  __shared__ float red[256];
  int r = blockIdx.x;
  if (r >= R) return;
  float s = 0.f;
  for (int c = threadIdx.x; c < C; c += blockDim.x) {
    float v = in[(long)r * C + c];
    s += v * v;
  }
  red[threadIdx.x] = s;
  __syncthreads();
  for (int st = 128; st > 0; st >>= 1) {
    if (threadIdx.x < st) red[threadIdx.x] += red[threadIdx.x + st];
    __syncthreads();
  }
  float rs = rsqrtf(red[0] + 1e-12f);
  for (int c = threadIdx.x; c < C; c += blockDim.x)
    out[(long)r * C + c] = in[(long)r * C + c] * rs;
}

// memory attention: logits = qn@mn^T/0.1, softmax, read = attn@mem ->
// channel-major second half of UCAT.
__global__ void mem_attn_k(const float* __restrict__ QN, const float* __restrict__ MN,
                           const float* __restrict__ MEM, float* __restrict__ UCAT,
                           int T, int C, int Mrows) {
  __shared__ float red[256];
  __shared__ float wgt[16];
  int t = blockIdx.x;
  for (int j = 0; j < Mrows; ++j) {
    float s = 0.f;
    for (int c = threadIdx.x; c < C; c += blockDim.x)
      s += QN[(long)t * C + c] * MN[(long)j * C + c];
    red[threadIdx.x] = s;
    __syncthreads();
    for (int st = 128; st > 0; st >>= 1) {
      if (threadIdx.x < st) red[threadIdx.x] += red[threadIdx.x + st];
      __syncthreads();
    }
    if (threadIdx.x == 0) wgt[j] = red[0] * 10.f;
    __syncthreads();
  }
  if (threadIdx.x == 0) {
    float mx = wgt[0];
    for (int j = 1; j < Mrows; ++j) mx = fmaxf(mx, wgt[j]);
    float sm = 0.f;
    for (int j = 0; j < Mrows; ++j) { wgt[j] = __expf(wgt[j] - mx); sm += wgt[j]; }
    float inv = 1.f / sm;
    for (int j = 0; j < Mrows; ++j) wgt[j] *= inv;
  }
  __syncthreads();
  for (int c = threadIdx.x; c < C; c += blockDim.x) {
    float r = 0.f;
    for (int j = 0; j < Mrows; ++j) r += wgt[j] * MEM[(long)j * C + c];
    UCAT[((long)(C + c)) * T + t] = r;
  }
}

__global__ void final_out_k(const float* __restrict__ XC, const float* __restrict__ O7,
                            float* __restrict__ out, int HW) {
  long idx = (long)blockIdx.x * blockDim.x + threadIdx.x;
  if (idx >= (long)HW * 7) return;
  int c = idx % 7; long t = idx / 7;
  out[idx] = ((c & 1) == 0) ? XC[(long)(c >> 1) * HW + t] : O7[(long)c * HW + t];
}

// ---------------------------------------------------------------------------
// Host orchestration
// ---------------------------------------------------------------------------
enum {
  IN_X = 0, IN_KEYS,
  HEAD_W1, HEAD_B1, HEAD_W2, HEAD_B2,
  CV1_LN_G, CV1_LN_B,
  CV1_SAG_W0, CV1_SAG_B0, CV1_SAG_W1, CV1_SAG_B1, CV1_SAG_W2, CV1_SAG_B2,
  CV1_COR_W0, CV1_COR_B0, CV1_COR_W1, CV1_COR_B1, CV1_COR_W2, CV1_COR_B2,
  CV2_LN_G, CV2_LN_B,
  CV2_SAG_W0, CV2_SAG_B0, CV2_SAG_W1, CV2_SAG_B1, CV2_SAG_W2, CV2_SAG_B2,
  CV2_COR_W0, CV2_COR_B0, CV2_COR_W1, CV2_COR_B1, CV2_COR_W2, CV2_COR_B2,
  ENC_W1, ENC_B1, ENC_W2, ENC_B2, ENC_W3, ENC_B3,
  P_MEM,
  DEC_W1, DEC_B1, DEC_W2, DEC_B2, DEC_W3, DEC_B3,
  IN_PROJ, CONV_W, CONV_B, X_PROJ, DT_W, DT_B, A_LOG, D_PARAM, OUT_PROJ,
  FUSE_W, FUSE_B, TAIL_W1, TAIL_B1, TAIL_W2, TAIL_B2
};

static const long HW = 65536, CHW = 64L * 65536;

static inline dim3 eg(long n) { return dim3((unsigned)((n + 255) / 256)); }

static void launch_conv(const float* X, const float* W, const float* B, float* out,
                        int Cin, int Cout, int Hin, int Win, int Hout, int Wout,
                        int ks, int stride, int pad, int trans, int act,
                        hipStream_t s) {
  dim3 g((Hout * Wout + 63) / 64, (Cout + 63) / 64);
  conv_wmma<<<g, 128, 0, s>>>(X, W, B, out, Cin, Cout, Hin, Win, Hout, Wout, ks,
                              stride, pad, trans, act);
}

static void launch_gemm(const float* W, const float* X, const float* B, float* out,
                        int M, int N, int K, int ldx, int act, hipStream_t s) {
  dim3 g((N + 63) / 64, (M + 63) / 64);
  gemm_xwT_wmma<<<g, 128, 0, s>>>(W, X, B, out, M, N, K, ldx, act);
}

static void run_crossview(const float* Xin, float* Yout, void* const* d_in, int base,
                          float* SCR, hipStream_t s) {
  float* XN = SCR;
  float* S1 = SCR + CHW;
  float* S2 = SCR + 2 * CHW;
  float* S3 = SCR + 3 * CHW;
  float* S4 = SCR + 4 * CHW;
  float* SAG = SCR + 5 * CHW;
  float* COR = SCR + 6 * CHW;
  layernorm_c_k<<<eg(HW), 256, 0, s>>>(Xin, (const float*)d_in[base],
                                       (const float*)d_in[base + 1], XN, 64, (int)HW);
  for (int br = 0; br < 2; ++br) {
    int wi = base + 2 + br * 6;
    int mode_in = (br == 0) ? 0 : 1;   // sag: swap01; cor: (2,0,1)
    int mode_out = (br == 0) ? 0 : 2;  // sag: swap01; cor: (1,2,0)
    float* OUTB = (br == 0) ? SAG : COR;
    launch_conv(XN, (const float*)d_in[wi], (const float*)d_in[wi + 1], S1,
                64, 64, 256, 256, 256, 256, 1, 1, 0, 0, 0, s);
    permute3_k<<<eg(CHW), 256, 0, s>>>(S1, S2, 64, 256, 256, mode_in);
    pixshuf_k<<<eg(CHW), 256, 0, s>>>(S2, S3, 256, 64, 256);     // -> (64,128,512)
    launch_conv(S3, (const float*)d_in[wi + 2], (const float*)d_in[wi + 3], S4,
                64, 64, 128, 512, 128, 512, 3, 1, 1, 0, 1, s);
    launch_conv(S4, (const float*)d_in[wi + 4], (const float*)d_in[wi + 5], S2,
                64, 64, 128, 512, 128, 512, 3, 1, 1, 0, 0, s);
    pixunshuf_k<<<eg(CHW), 256, 0, s>>>(S2, S3, 64, 128, 512);   // -> (256,64,256)
    permute3_k<<<eg(CHW), 256, 0, s>>>(S3, OUTB, 256, 64, 256, mode_out);
  }
  add2_k<<<eg(CHW), 256, 0, s>>>(SAG, COR, Yout, CHW);
}

static void run_mamba(const float* R1, float* OUT2, void* const* d_in, float* SCR,
                      hipStream_t s) {
  float* XZ = SCR;                  // 16384 x 1024
  float* U = XZ + 16777216;         // 16384 x 512
  float* DBL = U + 8388608;         // 16384 x 48
  float* DT = DBL + 786432;         // 16384 x 512 (dt, then y_mod in place)
  launch_gemm((const float*)d_in[IN_PROJ], R1, nullptr, XZ, 1024, 16384, 256, 256, 0, s);
  mamba_conv1d_k<<<eg(16384L * 512), 256, 0, s>>>(
      XZ, (const float*)d_in[CONV_W], (const float*)d_in[CONV_B], U, 64, 256, 512, 1024);
  launch_gemm((const float*)d_in[X_PROJ], U, nullptr, DBL, 48, 16384, 512, 512, 0, s);
  launch_gemm((const float*)d_in[DT_W], DBL, (const float*)d_in[DT_B], DT,
              512, 16384, 16, 48, 2, s);  // softplus
  mamba_scan_k<<<eg(64L * 512), 256, 0, s>>>(DBL, U, DT, XZ,
                                             (const float*)d_in[A_LOG],
                                             (const float*)d_in[D_PARAM], 64, 256);
  launch_gemm((const float*)d_in[OUT_PROJ], DT, nullptr, OUT2, 256, 16384, 512, 512, 0, s);
}

static void run_conv_ae(const float* R1, float* OUT1, void* const* d_in, float* SCR,
                        hipStream_t s) {
  float* F1 = SCR;                  // 128x128x128
  float* F2 = F1 + 2097152;         // 256x64x64
  float* FEA = F2 + 1048576;        // 512x32x32
  float* Q = FEA + 524288;          // 1024x512
  float* QN = Q + 524288;
  float* MN = QN + 524288;          // 10x512
  float* UC = MN + 8192;            // 1024x32x32
  float* D1 = UC + 1048576;         // 256x64x64
  float* D2 = D1 + 1048576;         // 128x128x128
  launch_conv(R1, (const float*)d_in[ENC_W1], (const float*)d_in[ENC_B1], F1,
              64, 128, 256, 256, 128, 128, 3, 2, 0, 0, 1, s);
  launch_conv(F1, (const float*)d_in[ENC_W2], (const float*)d_in[ENC_B2], F2,
              128, 256, 128, 128, 64, 64, 3, 2, 0, 0, 1, s);
  launch_conv(F2, (const float*)d_in[ENC_W3], (const float*)d_in[ENC_B3], FEA,
              256, 512, 64, 64, 32, 32, 3, 2, 0, 0, 1, s);
  chw_to_tc_k<<<eg(524288), 256, 0, s>>>(FEA, Q, 512, 1024);
  l2norm_rows_k<<<1024, 256, 0, s>>>(Q, QN, 1024, 512);
  l2norm_rows_k<<<10, 256, 0, s>>>((const float*)d_in[P_MEM], MN, 10, 512);
  hipMemcpyAsync(UC, FEA, 524288 * sizeof(float), hipMemcpyDeviceToDevice, s);
  mem_attn_k<<<1024, 256, 0, s>>>(QN, MN, (const float*)d_in[P_MEM], UC, 1024, 512, 10);
  launch_conv(UC, (const float*)d_in[DEC_W1], (const float*)d_in[DEC_B1], D1,
              1024, 256, 32, 32, 64, 64, 3, 2, 2, 1, 1, s);
  launch_conv(D1, (const float*)d_in[DEC_W2], (const float*)d_in[DEC_B2], D2,
              256, 128, 64, 64, 128, 128, 3, 2, 2, 1, 1, s);
  launch_conv(D2, (const float*)d_in[DEC_W3], (const float*)d_in[DEC_B3], OUT1,
              128, 64, 128, 128, 256, 256, 3, 2, 2, 1, 0, s);
}

extern "C" void kernel_launch(void* const* d_in, const int* in_sizes, int n_in,
                              void* d_out, int out_size, void* d_ws, size_t ws_size,
                              hipStream_t stream) {
  (void)in_sizes; (void)n_in; (void)out_size; (void)ws_size;
  float* ws = (float*)d_ws;
  float* XC = ws;                     // 4 x 256 x 256
  float* XHEAD = XC + 4 * HW;
  float* RES1 = XHEAD + CHW;
  float* OUT2 = RES1 + CHW;
  float* OUT1 = OUT2 + CHW;
  float* OUTP = OUT1 + CHW;           // "output", later reused as "res"
  float* RES2 = OUTP + CHW;
  float* O7 = RES2 + CHW;             // 7 x 256 x 256
  float* SCR = O7 + 7 * HW;           // stage scratch

  // xc = NHWC -> NCHW
  nhwc_to_nchw_k<<<eg(4 * HW), 256, 0, stream>>>((const float*)d_in[IN_X], XC, 256, 256, 4);

  // head
  float* H1 = SCR;
  launch_conv(XC, (const float*)d_in[HEAD_W1], (const float*)d_in[HEAD_B1], H1,
              4, 64, 256, 256, 256, 256, 3, 1, 1, 0, 1, stream);
  launch_conv(H1, (const float*)d_in[HEAD_W2], (const float*)d_in[HEAD_B2], XHEAD,
              64, 64, 256, 256, 256, 256, 3, 1, 1, 0, 0, stream);

  // res1 = crossview(x_head, cv1)
  run_crossview(XHEAD, RES1, d_in, CV1_LN_G, SCR, stream);

  // out2 = mamba(res1); out1 = conv_ae(res1)
  run_mamba(RES1, OUT2, d_in, SCR, stream);
  run_conv_ae(RES1, OUT1, d_in, SCR, stream);

  // output = out1 + out2 + res1
  add3_k<<<eg(CHW), 256, 0, stream>>>(OUT1, OUT2, RES1, OUTP, CHW);

  // res2 = crossview(output, cv2) + res1
  run_crossview(OUTP, RES2, d_in, CV2_LN_G, SCR, stream);
  add2_k<<<eg(CHW), 256, 0, stream>>>(RES2, RES1, RES2, CHW);

  // res = conv1x1(concat(res1,res2), fuse) + x_head + res1
  float* CAT = SCR;
  float* TMP = SCR + 2 * CHW;
  hipMemcpyAsync(CAT, RES1, CHW * sizeof(float), hipMemcpyDeviceToDevice, stream);
  hipMemcpyAsync(CAT + CHW, RES2, CHW * sizeof(float), hipMemcpyDeviceToDevice, stream);
  launch_conv(CAT, (const float*)d_in[FUSE_W], (const float*)d_in[FUSE_B], TMP,
              128, 64, 256, 256, 256, 256, 1, 1, 0, 0, 0, stream);
  add3_k<<<eg(CHW), 256, 0, stream>>>(TMP, XHEAD, RES1, OUTP, CHW);

  // tail
  float* O1 = SCR;
  launch_conv(OUTP, (const float*)d_in[TAIL_W1], (const float*)d_in[TAIL_B1], O1,
              64, 64, 256, 256, 256, 256, 3, 1, 1, 0, 1, stream);
  launch_conv(O1, (const float*)d_in[TAIL_W2], (const float*)d_in[TAIL_B2], O7,
              64, 7, 256, 256, 256, 256, 3, 1, 1, 0, 0, stream);

  // o[:, ::2] = xc; NCHW -> NHWC
  final_out_k<<<eg(7 * HW), 256, 0, stream>>>(XC, O7, (float*)d_out, (int)HW);
}
